// StaticQuantizedBMM1_85349590106240
// MI455X (gfx1250) — 8-bit matrix operand order repaired and checked by reading, compiled, not run on hardware
//
#include <hip/hip_runtime.h>

typedef int v8i __attribute__((ext_vector_type(8)));

#define S_DIM 2048
#define D_DIM 128
#define KC    64
#define TM    32

__device__ __forceinline__ int pack4(int a, int b, int c, int d) {
    return (a & 255) | ((b & 255) << 8) | ((c & 255) << 16) | ((d & 255) << 24);
}

__device__ __forceinline__ void quant1(float x, float inv, int& q, int& ind) {
    const float r = rintf(x * inv);
    ind = (r < -128.f) ? 1 : 0;
    q   = (int)fminf(fmaxf(r, -128.f), 127.f);
}

__global__ __launch_bounds__(256)
void quant_value_kernel(const float* __restrict__ value,
                        const float* __restrict__ kscale,
                        char* __restrict__ vq, char* __restrict__ vind,
                        int total_dwords)
{
    const int idx = blockIdx.x * 256 + threadIdx.x;
    if (idx >= total_dwords) return;
    const float invk = 1.0f / kscale[0];

    const int d  = idx & (D_DIM - 1);
    const int kb = idx >> 7;
    const int bh = kb >> 9;
    const int k0 = (kb & 511) << 2;

    const float* src = value + (size_t)bh * S_DIM * D_DIM + d;
    int q[4], ind[4];
    #pragma unroll
    for (int j = 0; j < 4; ++j)
        quant1(src[(size_t)(k0 + j) * D_DIM], invk, q[j], ind[j]);

    const size_t o = (size_t)bh * D_DIM * S_DIM + (size_t)d * S_DIM + k0;
    *reinterpret_cast<int*>(vq   + o) = pack4(q[0], q[1], q[2], q[3]);
    *reinterpret_cast<int*>(vind + o) = pack4(ind[0], ind[1], ind[2], ind[3]);
}

__global__ __launch_bounds__(256, 2)
void sqbmm_main_kernel(const float* __restrict__ score,
                       const char* __restrict__ vq,
                       const char* __restrict__ vind,
                       const float* __restrict__ qscale,
                       const float* __restrict__ kscale,
                       float* __restrict__ out)
{
    __shared__ __align__(16) char lA[2][TM * KC];

    const int t     = threadIdx.x;
    const int lane  = t & 31;
    const int wave  = t >> 5;
    const int bh    = blockIdx.x >> 6;
    const int strip = blockIdx.x & 63;
    const int m0    = strip * TM;

    const float* sBase = score + (size_t)bh * S_DIM * S_DIM + (size_t)m0 * S_DIM;
    const char*  bQ    = vq    + (size_t)bh * D_DIM * S_DIM;
    const char*  bI    = vind  + (size_t)bh * D_DIM * S_DIM;
    float*       oBase = out   + (size_t)bh * S_DIM * D_DIM + (size_t)m0 * D_DIM;

    const float qs   = qscale[0];
    const float invq = 1.0f / qs;
    const float fsc  = qs * kscale[0];

    const int aRow = t >> 3;
    const int aK8  = (t & 7) << 3;

    const int n0   = wave * 16;
    const int nsub = lane & 15;
    const int half = lane >> 4;

    v8i acc0  = {0,0,0,0,0,0,0,0}, accC0 = {0,0,0,0,0,0,0,0};
    v8i acc1  = {0,0,0,0,0,0,0,0}, accC1 = {0,0,0,0,0,0,0,0};

    for (int kc = 0; kc < S_DIM; kc += KC) {
        const int p = (kc >> 6) & 1;

        {
            const float* src = sBase + (size_t)aRow * S_DIM + kc + aK8;
            const float4 f0 = *reinterpret_cast<const float4*>(src);
            const float4 f1 = *reinterpret_cast<const float4*>(src + 4);
            int2 w;
            w.x = pack4((int)fminf(fmaxf(rintf(f0.x * invq), -128.f), 127.f),
                        (int)fminf(fmaxf(rintf(f0.y * invq), -128.f), 127.f),
                        (int)fminf(fmaxf(rintf(f0.z * invq), -128.f), 127.f),
                        (int)fminf(fmaxf(rintf(f0.w * invq), -128.f), 127.f));
            w.y = pack4((int)fminf(fmaxf(rintf(f1.x * invq), -128.f), 127.f),
                        (int)fminf(fmaxf(rintf(f1.y * invq), -128.f), 127.f),
                        (int)fminf(fmaxf(rintf(f1.z * invq), -128.f), 127.f),
                        (int)fminf(fmaxf(rintf(f1.w * invq), -128.f), 127.f));
            *reinterpret_cast<int2*>(&lA[p][aRow * KC + aK8]) = w;
            if (kc + KC < S_DIM)
                __builtin_prefetch(src + KC, 0, 3);
        }

        __syncthreads();

        v8i b, bi;
        {
            const size_t off = (size_t)(n0 + nsub) * S_DIM + kc + half * 16;
            const int4 b03 = *reinterpret_cast<const int4*>(bQ + off);
            const int4 b47 = *reinterpret_cast<const int4*>(bQ + off + 32);
            b[0]=b03.x; b[1]=b03.y; b[2]=b03.z; b[3]=b03.w;
            b[4]=b47.x; b[5]=b47.y; b[6]=b47.z; b[7]=b47.w;
            const int4 c03 = *reinterpret_cast<const int4*>(bI + off);
            const int4 c47 = *reinterpret_cast<const int4*>(bI + off + 32);
            bi[0]=c03.x; bi[1]=c03.y; bi[2]=c03.z; bi[3]=c03.w;
            bi[4]=c47.x; bi[5]=c47.y; bi[6]=c47.z; bi[7]=c47.w;
        }

        v8i a0, a1;
        {
            const char* r0 = &lA[p][nsub * KC + half * 16];
            const char* r1 = &lA[p][(16 + nsub) * KC + half * 16];
            int2 x;
            x = *reinterpret_cast<const int2*>(r0 +  0); a0[0]=x.x; a0[1]=x.y;
            x = *reinterpret_cast<const int2*>(r0 +  8); a0[2]=x.x; a0[3]=x.y;
            x = *reinterpret_cast<const int2*>(r0 + 32); a0[4]=x.x; a0[5]=x.y;
            x = *reinterpret_cast<const int2*>(r0 + 40); a0[6]=x.x; a0[7]=x.y;
            x = *reinterpret_cast<const int2*>(r1 +  0); a1[0]=x.x; a1[1]=x.y;
            x = *reinterpret_cast<const int2*>(r1 +  8); a1[2]=x.x; a1[3]=x.y;
            x = *reinterpret_cast<const int2*>(r1 + 32); a1[4]=x.x; a1[5]=x.y;
            x = *reinterpret_cast<const int2*>(r1 + 40); a1[6]=x.x; a1[7]=x.y;
        }

        acc0  = __builtin_amdgcn_wmma_i32_16x16x64_iu8(true, a0, true,  b,  acc0,  false, false);
        accC0 = __builtin_amdgcn_wmma_i32_16x16x64_iu8(true, a0, false, bi, accC0, false, false);
        acc1  = __builtin_amdgcn_wmma_i32_16x16x64_iu8(true, a1, true,  b,  acc1,  false, false);
        accC1 = __builtin_amdgcn_wmma_i32_16x16x64_iu8(true, a1, false, bi, accC1, false, false);
    }

    #pragma unroll
    for (int j = 0; j < 8; ++j) {
        const int m = half * 8 + j;
        const int n = n0 + nsub;
        oBase[(size_t)m * D_DIM + n]        = (float)(acc0[j] - accC0[j]) * fsc;
        oBase[(size_t)(16 + m) * D_DIM + n] = (float)(acc1[j] - accC1[j]) * fsc;
    }
}

__global__ __launch_bounds__(256, 2)
void sqbmm_fallback_kernel(const float* __restrict__ score,
                           const float* __restrict__ value,
                           const float* __restrict__ qscale,
                           const float* __restrict__ kscale,
                           float* __restrict__ out)
{
    __shared__ __align__(16) char lA [2][16 * KC];
    __shared__ __align__(16) char lB [2][D_DIM * KC];
    __shared__ __align__(16) char lBi[2][D_DIM * KC];

    const int t     = threadIdx.x;
    const int lane  = t & 31;
    const int wave  = t >> 5;
    const int bh    = blockIdx.x >> 7;
    const int strip = blockIdx.x & 127;
    const int m0    = strip * 16;

    const float* sBase = score + (size_t)bh * S_DIM * S_DIM + (size_t)m0 * S_DIM;
    const float* vBase = value + (size_t)bh * S_DIM * D_DIM;
    float*       oBase = out   + (size_t)bh * S_DIM * D_DIM + (size_t)m0 * D_DIM;

    const float qs   = qscale[0];
    const float invq = 1.0f / qs;
    const float invk = 1.0f / kscale[0];
    const float fsc  = qs * kscale[0];

    const int aRow = t >> 4, aK4 = (t & 15) << 2;
    const int n0 = wave * 16, nsub = lane & 15, half = lane >> 4;

    v8i acc = {0,0,0,0,0,0,0,0}, accC = {0,0,0,0,0,0,0,0};

    for (int kc = 0; kc < S_DIM; kc += KC) {
        const int p = (kc >> 6) & 1;
        {
            const float* src = sBase + (size_t)aRow * S_DIM + kc + aK4;
            const float4 f = *reinterpret_cast<const float4*>(src);
            *reinterpret_cast<int*>(&lA[p][aRow * KC + aK4]) =
                pack4((int)fminf(fmaxf(rintf(f.x * invq), -128.f), 127.f),
                      (int)fminf(fmaxf(rintf(f.y * invq), -128.f), 127.f),
                      (int)fminf(fmaxf(rintf(f.z * invq), -128.f), 127.f),
                      (int)fminf(fmaxf(rintf(f.w * invq), -128.f), 127.f));
        }
        #pragma unroll
        for (int i = 0; i < 8; ++i) {
            const int idx = t + i * 256;
            const int d = idx & (D_DIM - 1);
            const int k0 = (idx >> 7) << 2;
            int q[4], ind[4];
            #pragma unroll
            for (int j = 0; j < 4; ++j)
                quant1(vBase[(size_t)(kc + k0 + j) * D_DIM + d], invk, q[j], ind[j]);
            *reinterpret_cast<int*>(&lB [p][d * KC + k0]) = pack4(q[0], q[1], q[2], q[3]);
            *reinterpret_cast<int*>(&lBi[p][d * KC + k0]) = pack4(ind[0], ind[1], ind[2], ind[3]);
        }
        __syncthreads();
        v8i a, b, bi;
        {
            const char* rp = &lA[p][nsub * KC + half * 16];
            int2 x;
            x = *reinterpret_cast<const int2*>(rp +  0); a[0]=x.x; a[1]=x.y;
            x = *reinterpret_cast<const int2*>(rp +  8); a[2]=x.x; a[3]=x.y;
            x = *reinterpret_cast<const int2*>(rp + 32); a[4]=x.x; a[5]=x.y;
            x = *reinterpret_cast<const int2*>(rp + 40); a[6]=x.x; a[7]=x.y;
        }
        {
            const int n = n0 + nsub;
            const int4 b03 = *reinterpret_cast<const int4*>(&lB[p][n * KC + half * 16]);
            const int4 b47 = *reinterpret_cast<const int4*>(&lB[p][n * KC + 32 + half * 16]);
            b[0]=b03.x; b[1]=b03.y; b[2]=b03.z; b[3]=b03.w;
            b[4]=b47.x; b[5]=b47.y; b[6]=b47.z; b[7]=b47.w;
            const int4 c03 = *reinterpret_cast<const int4*>(&lBi[p][n * KC + half * 16]);
            const int4 c47 = *reinterpret_cast<const int4*>(&lBi[p][n * KC + 32 + half * 16]);
            bi[0]=c03.x; bi[1]=c03.y; bi[2]=c03.z; bi[3]=c03.w;
            bi[4]=c47.x; bi[5]=c47.y; bi[6]=c47.z; bi[7]=c47.w;
        }
        acc  = __builtin_amdgcn_wmma_i32_16x16x64_iu8(true, a, true,  b,  acc,  false, false);
        accC = __builtin_amdgcn_wmma_i32_16x16x64_iu8(true, a, false, bi, accC, false, false);
    }
    #pragma unroll
    for (int j = 0; j < 8; ++j)
        oBase[(size_t)(half * 8 + j) * D_DIM + n0 + nsub] = (float)(acc[j] - accC[j]) * fsc;
}

extern "C" void kernel_launch(void* const* d_in, const int* in_sizes, int n_in,
                              void* d_out, int out_size, void* d_ws, size_t ws_size,
                              hipStream_t stream) {
    (void)n_in; (void)out_size;
    const float* score = (const float*)d_in[0];
    const float* value = (const float*)d_in[1];
    const float* qsc   = (const float*)d_in[2];
    const float* ksc   = (const float*)d_in[3];
    float* out = (float*)d_out;

    const int BH = in_sizes[1] / (S_DIM * D_DIM);
    const size_t plane = (size_t)BH * D_DIM * S_DIM;

    if (ws_size >= 2 * plane) {
        char* vq   = (char*)d_ws;
        char* vind = (char*)d_ws + plane;
        const int total_dwords = (int)(plane / 4);
        quant_value_kernel<<<(total_dwords + 255) / 256, 256, 0, stream>>>(
            value, ksc, vq, vind, total_dwords);
        sqbmm_main_kernel<<<dim3((unsigned)(BH * (S_DIM / TM))), 256, 0, stream>>>(
            score, vq, vind, qsc, ksc, out);
    } else {
        sqbmm_fallback_kernel<<<dim3((unsigned)(BH * (S_DIM / 16))), 256, 0, stream>>>(
            score, value, qsc, ksc, out);
    }
}
